// DeepFatorizationMachine_53961969107175
// MI455X (gfx1250) — matrix operand order repaired and checked by reading, not compiled, not run on hardware
//
#include <hip/hip_runtime.h>
#include <hip/hip_bf16.h>
#include <math.h>

// ---------------------------------------------------------------------------
// DeepFM fused inference for MI455X (gfx1250, wave32, WMMA).
//   out = sigmoid( linear + 0.5*mean((xK)^2 - x^2K^2) + MLP(emb) )
// One workgroup = 256 threads = 8 wave32 = 128 batch rows.
// All GEMMs on v_wmma_f32_16x16x32_f16 (f16 in, f32 accumulate).
// LDS data paths are fully vectorized: every WMMA fragment is 2x ds_load_b128.
// ---------------------------------------------------------------------------

#define HASH_BINS 100000
#define EMB 64

typedef __attribute__((ext_vector_type(16))) _Float16 v16h;
typedef __attribute__((ext_vector_type(8)))  _Float16 v8h;
typedef __attribute__((ext_vector_type(4)))  _Float16 v4h;
typedef __attribute__((ext_vector_type(8)))  float    v8f;

#define TM        128   // batch rows per workgroup
#define NTHREADS  256   // 8 waves
#define HSTRIDE   256   // activation buffer stride (halves)

// ---- LDS partition (bytes) ----
#define OFF_E     0                       // 128x128 f16 = 32768
#define OFF_H0    32768                   // 128x256 f16 = 65536
#define OFF_H1    (OFF_H0 + 65536)        // 128x256 f16 = 65536
#define OFF_W     (OFF_H1 + 65536)        // up to 224x256 f16 = 114688
#define OFF_CROSS (OFF_W + 114688)        // 128 f32
#define OFF_LIN   (OFF_CROSS + 512)       // 128 f32
#define SMEM_BYTES (OFF_LIN + 512)        // = 279552 (< 320KB/WGP)

static __device__ inline v8f wmma_f16(v16h a, v16h b, v8f c) {
  return __builtin_amdgcn_wmma_f32_16x16x32_f16(
      /*neg_a=*/false, a, /*neg_b=*/false, b,
      /*c_mod=*/(short)0, c, /*reuse_a=*/false, /*reuse_b=*/false);
}

// XOR bank swizzle for activations, 8-half (16B) chunk granularity:
// element (row, col) lives at row*STRIDE + ((col>>3 ^ (row&7))<<3 | (col&7)).
static __device__ inline int swz(int row, int col) {
  return ((((col >> 3) ^ (row & 7)) << 3) | (col & 7));
}

// A fragment: rows [row0,row0+16) x K [k0,k0+32) from swizzled row-major LDS.
// CDNA5 16-bit A layout: lane holds row M=lane&15; half h holds
//   K = k0 + 16*(h>=8) + 8*(lane>=16) + (h&7)
// -> two contiguous 8-half runs per lane -> 2x ds_load_b128.
template <int STRIDE>
static __device__ inline v16h load_a_frag(const _Float16* A, int row0, int k0, int lane) {
  const int m   = lane & 15;
  const int hi  = lane >> 4;
  const int row = row0 + m;
  const int sm  = row & 7;
  const v8h* base = (const v8h*)(A + row * STRIDE);
  const int c0 = (k0 >> 3) + hi;       // chunk for halves 0..7
  v8h lo = base[c0 ^ sm];
  v8h hv = base[(c0 + 2) ^ sm];        // chunk for halves 8..15 (K+16)
  return __builtin_shufflevector(lo, hv, 0, 1, 2, 3, 4, 5, 6, 7,
                                         8, 9, 10, 11, 12, 13, 14, 15);
}

// B fragments are staged pre-packed: tile (kt,nt) of the KxN weight matrix is
// stored as [tile][half_sel s][lane][8 halves]; lane = (n&15) | ((k>>3)&1)<<4,
// halves hold K = kt*32 + 16*(h>=8) + 8*(lane>=16) + (h&7) (h=0..15, s=h>>3).
// -> 2x ds_load_b128, 16B per-lane stride => conflict-free.
template <int NPAD>
static __device__ inline v16h load_b_frag(const _Float16* W, int nt, int kt, int lane) {
  constexpr int NT = NPAD / 16;
  const int tile = kt * NT + nt;
  const v8h* p = (const v8h*)W;
  v8h lo = p[((tile << 1) | 0) * 32 + lane];
  v8h hv = p[((tile << 1) | 1) * 32 + lane];
  return __builtin_shufflevector(lo, hv, 0, 1, 2, 3, 4, 5, 6, 7,
                                         8, 9, 10, 11, 12, 13, 14, 15);
}

// Stage Kreal x Nreal row-major f32 weights into fragment-packed f16 LDS,
// zero-filling K/N padding so padded tiles contribute nothing.
template <int KPAD, int NPAD>
static __device__ inline void stage_w_frag(_Float16* dst, const float* src,
                                           int Kreal, int Nreal, int tid) {
  constexpr int NT = NPAD / 16;
  const int total = KPAD * NPAD;
  for (int idx = tid; idx < total; idx += NTHREADS) {
    const int k = idx / NPAD;
    const int n = idx - k * NPAD;
    float v = 0.0f;
    if (k < Kreal && n < Nreal) v = src[k * Nreal + n];
    const int kt = k >> 5, nt = n >> 4;
    const int kk = k & 31;
    const int hi = (kk >> 3) & 1;
    const int h  = (kk & 7) | ((kk >> 4) << 3);
    const int lane = (n & 15) | (hi << 4);
    const int tile = kt * NT + nt;
    dst[(((tile << 1) | (h >> 3)) << 8) + (lane << 3) + (h & 7)] = (_Float16)v;
  }
}

static __device__ inline void zero_h(_Float16* h, int tid) {
  v8h z = {};
  v8h* p = (v8h*)h;
  for (int idx = tid; idx < TM * HSTRIDE / 8; idx += NTHREADS) p[idx] = z;
}

// Pull a weight array toward this WGP while other work proceeds.
static __device__ inline void prefetch_bytes(const void* p, int bytes, int tid) {
  const char* c = (const char*)p;
  for (int off = tid * 128; off < bytes; off += NTHREADS * 128)
    __builtin_prefetch(c + off, 0, 1);  // global_prefetch_b8
}

// One dense layer: Hout[TM x NPAD] = act(A[TM x KPAD] @ W[KPAD x NPAD] + bias)
template <int KPAD, int NPAD, int NREAL, int ASTRIDE, bool RELU>
static __device__ inline void gemm_layer(const _Float16* A, const _Float16* W,
                                         const float* bias, _Float16* Hout,
                                         int wave, int lane) {
  const int row0 = wave * 16;
  const int hi = lane >> 4;
  const int nl = lane & 15;
#pragma unroll 1
  for (int nt = 0; nt < NPAD / 16; ++nt) {
    v8f c = {};
#pragma unroll
    for (int kt = 0; kt < KPAD / 32; ++kt) {
      v16h a = load_a_frag<ASTRIDE>(A, row0, kt * 32, lane);
      v16h b = load_b_frag<NPAD>(W, nt, kt, lane);
      c = wmma_f16(a, b, c);
    }
    const int n = nt * 16 + nl;
    const float bv = (n < NREAL) ? bias[n] : 0.0f;
    // C layout: lanes 0-15 hold M=v, lanes 16-31 hold M=v+8, col = n.
#pragma unroll
    for (int v = 0; v < 8; ++v) {
      const int row = row0 + v + hi * 8;
      float val = c[v] + bv;
      if (RELU) val = fmaxf(val, 0.0f);
      Hout[row * HSTRIDE + swz(row, n)] = (_Float16)val;
    }
  }
}

__global__ void __launch_bounds__(NTHREADS)
deepfm_kernel(const int* __restrict__ user_idx, const int* __restrict__ item_idx,
              const float* __restrict__ user_emb, const float* __restrict__ item_emb,
              const float* __restrict__ lin_w, const float* __restrict__ lin_b,
              const float* __restrict__ cross_k,
              const float* __restrict__ W1, const float* __restrict__ b1,
              const float* __restrict__ W2, const float* __restrict__ b2,
              const float* __restrict__ W3, const float* __restrict__ b3,
              const float* __restrict__ W4, const float* __restrict__ b4,
              const float* __restrict__ W5, const float* __restrict__ b5,
              float* __restrict__ out, int nB) {
  extern __shared__ __align__(16) char smem[];
  _Float16* smE  = (_Float16*)(smem + OFF_E);
  _Float16* smH0 = (_Float16*)(smem + OFF_H0);
  _Float16* smH1 = (_Float16*)(smem + OFF_H1);
  _Float16* smW  = (_Float16*)(smem + OFF_W);
  float* smCross = (float*)(smem + OFF_CROSS);
  float* smLin   = (float*)(smem + OFF_LIN);

  const int tid  = threadIdx.x;
  const int wave = tid >> 5;
  const int lane = tid & 31;
  const int rowBase = blockIdx.x * TM;

  // ---- Stage 1: gather embeddings -> smE (f16, swizzled) with float4 loads,
  //               linear term, zero cross acc, stage cross_k -> smW. ----
  for (int u = tid; u < TM * 32; u += NTHREADS) {   // 32 float4 chunks per row
    const int r = u >> 5;
    const int q = u & 31;
    int grow = rowBase + r;
    if (grow >= nB) grow = 0;  // safe clamp (nB is a multiple of TM in practice)
    float4 f;
    if (q < 16) f = ((const float4*)user_emb)[(long)user_idx[grow] * 16 + q];
    else        f = ((const float4*)item_emb)[(long)item_idx[grow] * 16 + (q - 16)];
    const int col = q * 4;
    v4h hv = {(_Float16)f.x, (_Float16)f.y, (_Float16)f.z, (_Float16)f.w};
    *(v4h*)(smE + r * 128 + swz(r, col)) = hv;
  }
  if (tid < TM) {
    int grow = rowBase + tid;
    if (grow >= nB) grow = 0;
    smLin[tid] = lin_w[user_idx[grow]] + lin_w[HASH_BINS + item_idx[grow]] + lin_b[0];
    smCross[tid] = 0.0f;
  }
  stage_w_frag<128, 128>(smW, cross_k, 2 * EMB, 2 * EMB, tid);
  prefetch_bytes(W1, 128 * 200 * 4, tid);
  __syncthreads();

  // ---- Stage 2: FM cross term. XK = E@K ; X2K2 = E^2 @ K^2 (squares formed
  //      in-register on f16 fragments). acc[v] = sum_n(xk^2 - x2k2) per row,
  //      reduced across the 16 column-lanes with one ds_add_f32 per row/lane. ----
  {
    const int row0 = wave * 16;
    const int hi = lane >> 4;
    float acc[8];
#pragma unroll
    for (int v = 0; v < 8; ++v) acc[v] = 0.0f;
#pragma unroll 1
    for (int nt = 0; nt < 8; ++nt) {
      v8f cx = {};
      v8f c2 = {};
#pragma unroll
      for (int kt = 0; kt < 4; ++kt) {
        v16h a = load_a_frag<128>(smE, row0, kt * 32, lane);
        v16h b = load_b_frag<128>(smW, nt, kt, lane);
        v16h a2, b2;
#pragma unroll
        for (int h = 0; h < 16; ++h) { a2[h] = a[h] * a[h]; b2[h] = b[h] * b[h]; }
        cx = wmma_f16(a, b, cx);
        c2 = wmma_f16(a2, b2, c2);
      }
#pragma unroll
      for (int v = 0; v < 8; ++v) acc[v] += cx[v] * cx[v] - c2[v];
    }
#pragma unroll
    for (int v = 0; v < 8; ++v)
      atomicAdd(&smCross[row0 + v + hi * 8], acc[v]);  // ds_add_f32
  }
  __syncthreads();

  // ---- Layer 1: H0 = relu(E @ W1 + b1)   [K 128, N 200 -> 208 pad] ----
  stage_w_frag<128, 208>(smW, W1, 128, 200, tid);
  zero_h(smH0, tid);
  prefetch_bytes(W2, 200 * 256 * 4, tid);
  __syncthreads();
  gemm_layer<128, 208, 200, 128, true>(smE, smW, b1, smH0, wave, lane);
  __syncthreads();

  // ---- Layer 2: H1 = relu(H0 @ W2 + b2)  [K 200 -> 224 pad, N 256] ----
  stage_w_frag<224, 256>(smW, W2, 200, 256, tid);
  zero_h(smH1, tid);
  prefetch_bytes(W3, 256 * 200 * 4, tid);
  __syncthreads();
  gemm_layer<224, 256, 256, HSTRIDE, true>(smH0, smW, b2, smH1, wave, lane);
  __syncthreads();

  // ---- Layer 3: H0 = relu(H1 @ W3 + b3)  [K 256, N 200 -> 208 pad] ----
  stage_w_frag<256, 208>(smW, W3, 256, 200, tid);
  zero_h(smH0, tid);
  prefetch_bytes(W4, 200 * 128 * 4, tid);
  __syncthreads();
  gemm_layer<256, 208, 200, HSTRIDE, true>(smH1, smW, b3, smH0, wave, lane);
  __syncthreads();

  // ---- Layer 4: H1 = relu(H0 @ W4 + b4)  [K 200 -> 224 pad, N 128] ----
  stage_w_frag<224, 128>(smW, W4, 200, 128, tid);
  zero_h(smH1, tid);
  __syncthreads();
  gemm_layer<224, 128, 128, HSTRIDE, true>(smH0, smW, b4, smH1, wave, lane);
  __syncthreads();

  // ---- Layer 5 + combine: dnn = H1 @ W5 + b5 ; out = sigmoid(lin+cross+dnn).
  //      Read H1 row in swizzled 8-half chunks (ds_load_b128). ----
  if (tid < TM) {
    const int grow = rowBase + tid;
    const v8h* hb = (const v8h*)(smH1 + tid * HSTRIDE);
    const int sm = tid & 7;
    float acc = 0.0f;
#pragma unroll 1
    for (int cj = 0; cj < 16; ++cj) {
      v8h hc = hb[cj ^ sm];
#pragma unroll
      for (int e = 0; e < 8; ++e) acc = fmaf((float)hc[e], W5[cj * 8 + e], acc);
    }
    const float dnn = acc + b5[0];
    const float x = smLin[tid] + (0.5f / 128.0f) * smCross[tid] + dnn;
    if (grow < nB) out[grow] = 1.0f / (1.0f + __expf(-x));
  }
}

extern "C" void kernel_launch(void* const* d_in, const int* in_sizes, int n_in,
                              void* d_out, int out_size, void* d_ws, size_t ws_size,
                              hipStream_t stream) {
  const int*   user_idx = (const int*)d_in[0];
  const int*   item_idx = (const int*)d_in[1];
  const float* user_emb = (const float*)d_in[2];
  const float* item_emb = (const float*)d_in[3];
  const float* lin_w    = (const float*)d_in[4];
  const float* lin_b    = (const float*)d_in[5];
  const float* cross_k  = (const float*)d_in[6];
  const float* W1 = (const float*)d_in[7];  const float* b1 = (const float*)d_in[8];
  const float* W2 = (const float*)d_in[9];  const float* b2 = (const float*)d_in[10];
  const float* W3 = (const float*)d_in[11]; const float* b3 = (const float*)d_in[12];
  const float* W4 = (const float*)d_in[13]; const float* b4 = (const float*)d_in[14];
  const float* W5 = (const float*)d_in[15]; const float* b5 = (const float*)d_in[16];
  float* out = (float*)d_out;

  const int nB = in_sizes[0];
  const int grid = (nB + TM - 1) / TM;

  // Allow large dynamic LDS (320 KB/WGP on gfx1250). Deterministic, capture-safe.
  (void)hipFuncSetAttribute(reinterpret_cast<const void*>(&deepfm_kernel),
                            hipFuncAttributeMaxDynamicSharedMemorySize, SMEM_BYTES);

  deepfm_kernel<<<grid, NTHREADS, SMEM_BYTES, stream>>>(
      user_idx, item_idx, user_emb, item_emb, lin_w, lin_b, cross_k,
      W1, b1, W2, b2, W3, b3, W4, b4, W5, b5, out, nB);
}
